// LNN_27032524161660
// MI455X (gfx1250) — compile-verified, matrix operand order checked by reading, not run on hardware
//
#include <hip/hip_runtime.h>
#include <hip/hip_bf16.h>
#include <cstdint>

// ---------------------------------------------------------------------------
// LNN forward for MI455X (gfx1250, wave32, WMMA).
//   x   : [2048, 64, 512]  f32
//   Wf/Wg/Wh : [1024, 1536] f32, b* : [1024]
//   Wfc : [2, 1024] f32, bfc : [2]
//   out : [2048, 2] f32
//
// Persistent workgroups: each owns 16 batch rows for all 64 time steps.
// combined = [x_t | h] kept in LDS as bf16 (16 x 1536, padded pitch).
// Weights pre-packed into d_ws as bf16 WMMA B-fragments (32B/lane, coalesced).
// Inner loop: issue all 12 fragment loads, then 12 WMMAs -> loads overlap
// compute via decreasing s_wait_loadcnt thresholds instead of serializing.
// ---------------------------------------------------------------------------

typedef __attribute__((ext_vector_type(16))) __bf16 v16bf;
typedef __attribute__((ext_vector_type(8)))  float  v8f;

#define BATCH   2048
#define TLEN    64
#define DIN     512
#define HID     1024
#define CIN     1536              // DIN + HID
#define MT      16                // batch rows per workgroup
#define NWAVES  16
#define NTHREADS (NWAVES * 32)
#define KSTEPS  (CIN / 32)        // 48 K-steps of 32
#define QT      4                 // column-tiles (of 16) per wave: 64/NWAVES
#define PITCH   1544              // bf16 elems per LDS row (1536 + 8 pad)
#define PFDIST  6                 // prefetch distance in K-steps

// Packed-weight element count: 64 colTiles * 3 mats * 48 ksteps * 32 lanes * 16
#define PACK_ELEMS (64 * 3 * KSTEPS * 32 * 16)

// K index inside a 32-wide K block for (lane, elem) per the 16-bit operand
// VGPR layout (ISA 7.12.2), B mirrored from A with lanes = N:
//   lanes 0-15 : elems 0-7 -> K 0-7,  elems 8-15 -> K 16-23
//   lanes16-31 : elems 0-7 -> K 8-15, elems 8-15 -> K 24-31
__device__ __host__ inline int frag_k(int lane, int e) {
    return (e & 7) + ((e >> 3) << 4) + (((lane >> 4) & 1) << 3);
}

// ---------------------------------------------------------------------------
// Kernel 1: pack Wf/Wg/Wh (f32 row-major [1024,1536]) into bf16 B-fragments.
// Layout: idx = (((jt*3 + s)*KSTEPS + kk)*32 + lane)*16 + e
// ---------------------------------------------------------------------------
__global__ void lnn_pack_weights(const float* __restrict__ Wf,
                                 const float* __restrict__ Wg,
                                 const float* __restrict__ Wh,
                                 __bf16* __restrict__ pack) {
    int idx = blockIdx.x * blockDim.x + threadIdx.x;
    if (idx >= PACK_ELEMS) return;
    int e    = idx & 15;
    int lane = (idx >> 4) & 31;
    int kk   = (idx >> 9) % KSTEPS;
    int rest = (idx >> 9) / KSTEPS;   // jt*3 + s
    int s    = rest % 3;
    int jt   = rest / 3;
    int k = 32 * kk + frag_k(lane, e);
    int n = jt * 16 + (lane & 15);
    const float* W = (s == 0) ? Wf : ((s == 1) ? Wg : Wh);
    pack[idx] = (__bf16)W[n * CIN + k];
}

// ---------------------------------------------------------------------------
// Kernel 2: persistent recurrence + final FC.
// ---------------------------------------------------------------------------
__global__ __launch_bounds__(NTHREADS)
void lnn_forward(const float* __restrict__ x,
                 const float* __restrict__ bfv,
                 const float* __restrict__ bgv,
                 const float* __restrict__ bhv,
                 const __bf16* __restrict__ wpack,
                 const float* __restrict__ Wfc,
                 const float* __restrict__ bfc,
                 float* __restrict__ out) {
    __shared__ __bf16 sComb[MT * PITCH];   // [16 rows][512 x | 1024 h], bf16

    const int tid  = threadIdx.x;
    const int lane = tid & 31;
    const int wave = tid >> 5;
    const int b0   = blockIdx.x * MT;

    const int laneGroup = lane >> 4;   // 0: M/N half 0-7, 1: half 8-15
    const int row16     = lane & 15;

    // Per-lane base into packed weights: wave's first fragment, this lane.
    // Fragment (q,s) at K-step kk lives at byte offset:
    //   ((q*3+s)*KSTEPS)*1024 (immediate) + kk*1024 (pointer walk)
    const __bf16* wl = wpack + ((size_t)(wave * QT * 3) * KSTEPS) * 512
                             + (size_t)lane * 16;

    // ---- init: h = 0 in LDS ----
    for (int i = tid; i < MT * HID; i += NTHREADS) {
        int r = i >> 10, c = i & (HID - 1);
        sComb[r * PITCH + DIN + c] = (__bf16)0.0f;
    }
    // ---- load x_0 (f32 -> bf16 -> LDS) : 32 threads/row, 16 floats each ----
    {
        int tr = tid >> 5, tc = tid & 31;
        const float4* xp = reinterpret_cast<const float4*>(
            x + (((size_t)(b0 + tr)) * TLEN + 0) * DIN) + tc * 4;
        __bf16* dst = sComb + tr * PITCH + tc * 16;
#pragma unroll
        for (int i = 0; i < 4; ++i) {
            float4 v = xp[i];
            dst[i * 4 + 0] = (__bf16)v.x; dst[i * 4 + 1] = (__bf16)v.y;
            dst[i * 4 + 2] = (__bf16)v.z; dst[i * 4 + 3] = (__bf16)v.w;
        }
    }
    __syncthreads();

    // ---- per-lane biases for the wave's 4 column-tiles ----
    float biasF[QT], biasG[QT], biasH[QT];
#pragma unroll
    for (int q = 0; q < QT; ++q) {
        int n = (wave * QT + q) * 16 + row16;
        biasF[q] = bfv[n]; biasG[q] = bgv[n]; biasH[q] = bhv[n];
    }

    // ---- recurrence over 64 time steps ----
    for (int t = 0; t < TLEN; ++t) {
        v8f acc[QT][3];
#pragma unroll
        for (int q = 0; q < QT; ++q)
#pragma unroll
            for (int s = 0; s < 3; ++s)
                acc[q][s] = (v8f){0.f, 0.f, 0.f, 0.f, 0.f, 0.f, 0.f, 0.f};

        const __bf16* wk = wl;   // walks by 512 elems (1 KB) per K-step

#pragma unroll 1
        for (int kk = 0; kk < KSTEPS; ++kk, wk += 512) {
            // ---- issue ALL 12 B-fragment loads first (24 b128, clause) ----
            v16bf bfr[QT][3];
#pragma unroll
            for (int q = 0; q < QT; ++q)
#pragma unroll
                for (int s = 0; s < 3; ++s)
                    bfr[q][s] = *reinterpret_cast<const v16bf*>(
                        wk + (size_t)((q * 3 + s) * KSTEPS) * 512);

            // ---- A-fragment from LDS: row = lane&15, two 16B chunks ----
            union { v16bf v; uint4 u[2]; } afrag;
            const char* abase = reinterpret_cast<const char*>(sComb)
                              + row16 * (PITCH * 2) + kk * 64 + laneGroup * 16;
            afrag.u[0] = *reinterpret_cast<const uint4*>(abase);
            afrag.u[1] = *reinterpret_cast<const uint4*>(abase + 32);

            // ---- prefetch fragments PFDIST K-steps ahead ----
            if (kk + PFDIST < KSTEPS) {
#pragma unroll
                for (int q = 0; q < QT; ++q)
#pragma unroll
                    for (int s = 0; s < 3; ++s)
                        __builtin_prefetch(
                            wk + (size_t)PFDIST * 512
                               + (size_t)((q * 3 + s) * KSTEPS) * 512, 0, 0);
            }

            // ---- 12 WMMAs; waits resolve per-fragment, loads stay in flight
#pragma unroll
            for (int q = 0; q < QT; ++q)
#pragma unroll
                for (int s = 0; s < 3; ++s)
                    acc[q][s] = __builtin_amdgcn_wmma_f32_16x16x32_bf16(
                        false, afrag.v, false, bfr[q][s],
                        (short)0, acc[q][s], false, false);
        }

        // ---- bias + gate: new_h = h_out + sigmoid(-f) * (g - h_out) ----
        float nh[QT][8];
#pragma unroll
        for (int q = 0; q < QT; ++q)
#pragma unroll
            for (int r = 0; r < 8; ++r) {
                float f  = acc[q][0][r] + biasF[q];
                float g  = acc[q][1][r] + biasG[q];
                float hh = acc[q][2][r] + biasH[q];
                float gate = 1.0f / (1.0f + __expf(f));   // sigmoid(-f)
                nh[q][r] = hh + gate * (g - hh);
            }

        __syncthreads();   // all LDS reads of old combined are done

        // store new h (bf16) into LDS h region; C layout: lane->(n, m-half)
#pragma unroll
        for (int q = 0; q < QT; ++q) {
            int n = (wave * QT + q) * 16 + row16;
#pragma unroll
            for (int r = 0; r < 8; ++r) {
                int m = r + 8 * laneGroup;
                sComb[m * PITCH + DIN + n] = (__bf16)nh[q][r];
            }
        }
        // load x_{t+1}
        if (t + 1 < TLEN) {
            int tr = tid >> 5, tc = tid & 31;
            const float4* xp = reinterpret_cast<const float4*>(
                x + (((size_t)(b0 + tr)) * TLEN + (t + 1)) * DIN) + tc * 4;
            __bf16* dst = sComb + tr * PITCH + tc * 16;
#pragma unroll
            for (int i = 0; i < 4; ++i) {
                float4 v = xp[i];
                dst[i * 4 + 0] = (__bf16)v.x; dst[i * 4 + 1] = (__bf16)v.y;
                dst[i * 4 + 2] = (__bf16)v.z; dst[i * 4 + 3] = (__bf16)v.w;
            }
        }
        __syncthreads();
    }

    // ---- final FC: out[b, 0..1] = h_final . Wfc[o,:] + bfc[o] ----
    if (tid < MT * 2) {
        int r = tid >> 1, o = tid & 1;
        float sum = bfc[o];
        const float* w = Wfc + o * HID;
        for (int k = 0; k < HID; ++k)
            sum += (float)sComb[r * PITCH + DIN + k] * w[k];
        out[(b0 + r) * 2 + o] = sum;
    }
}

// ---------------------------------------------------------------------------
extern "C" void kernel_launch(void* const* d_in, const int* in_sizes, int n_in,
                              void* d_out, int out_size, void* d_ws, size_t ws_size,
                              hipStream_t stream) {
    const float* x   = (const float*)d_in[0];
    const float* Wf  = (const float*)d_in[1];
    const float* bf_ = (const float*)d_in[2];
    const float* Wg  = (const float*)d_in[3];
    const float* bg_ = (const float*)d_in[4];
    const float* Wh  = (const float*)d_in[5];
    const float* bh_ = (const float*)d_in[6];
    const float* Wfc = (const float*)d_in[7];
    const float* bfc = (const float*)d_in[8];
    float* out = (float*)d_out;

    __bf16* wpack = (__bf16*)d_ws;   // needs 3*1024*1536*2 B = 9.44 MB

    lnn_pack_weights<<<(PACK_ELEMS + 255) / 256, 256, 0, stream>>>(
        Wf, Wg, Wh, wpack);

    lnn_forward<<<BATCH / MT, NTHREADS, 0, stream>>>(
        x, bf_, bg_, bh_, wpack, Wfc, bfc, out);
}
